// DeformableLayer_64269890617665
// MI455X (gfx1250) — matrix operand order repaired and checked by reading, not compiled, not run on hardware
//
#include <hip/hip_runtime.h>
#include <math.h>

// ---------------- constants ----------------
#define Hh 128
#define Ww 128
#define CIN 64
#define COUT 128
#define KDIM 576            // Cin * 9
#define NPIX 131072         // B * H * W
#define TN 32               // pixels per workgroup (both GEMM kernels)
#define LDSTR 584           // bf16 elems per pixel row in LDS (576 + 8 pad)

typedef __attribute__((ext_vector_type(16))) __bf16 v16bf;
typedef __attribute__((ext_vector_type(8)))  float  v8f;

__device__ __forceinline__ unsigned short f2bf(float f) {
  unsigned int u = __float_as_uint(f);
  unsigned int r = u + 0x7FFFu + ((u >> 16) & 1u);   // round-to-nearest-even
  return (unsigned short)(r >> 16);
}

// ---------------- k0a: conv_w fp32 -> bf16 [128][576] ----------------
__global__ void k_cvt_w(const float* __restrict__ w, unsigned short* __restrict__ wb) {
  int i = blockIdx.x * 256 + threadIdx.x;
  if (i < COUT * KDIM) wb[i] = f2bf(w[i]);
}

// ---------------- k0b: offset_w(18)+mask_w(9)+zero(5) -> bf16 [32][576] ----------------
__global__ void k_cvt_om(const float* __restrict__ ow, const float* __restrict__ mw,
                         unsigned short* __restrict__ wb2) {
  int i = blockIdx.x * 256 + threadIdx.x;
  if (i >= 32 * KDIM) return;
  int j = i / KDIM, r = i % KDIM;
  float v = (j < 18) ? ow[j * KDIM + r] : (j < 27 ? mw[(j - 18) * KDIM + r] : 0.f);
  wb2[i] = f2bf(v);
}

// ---------------- k1: offset+mask 3x3 conv as bf16 WMMA implicit GEMM ----------------
// M=32 (27 used), K=576, N=32 pixels per workgroup; 128 threads = 4 waves,
// wave -> one 16x16 C-tile (mtile = wv>>1, ntile = wv&1).
__global__ void __launch_bounds__(128) k_conv_om(
    const float* __restrict__ x, const unsigned short* __restrict__ wb2,
    const float* __restrict__ ob, const float* __restrict__ mb,
    float* __restrict__ offs, float* __restrict__ maskb) {
  __shared__ unsigned short Vlds[TN * LDSTR];

  int tid  = threadIdx.x;
  int pix0 = blockIdx.x * TN;

  // ---- phase 1: im2col 32 pixels x 576 (zero-padded 3x3 patches) into LDS bf16 ----
  {
    int p = tid >> 2;            // 0..31 pixel in tile
    int s = tid & 3;             // 4 threads/pixel -> 16 channels each
    int g = pix0 + p;
    int b = g >> 14, hw = g & 16383;
    int h = hw >> 7, w = hw & 127;
    const float* xb = x + (size_t)b * CIN * (Hh * Ww);
    unsigned short* vrow = &Vlds[p * LDSTR];
    for (int dydx = 0; dydx < 9; ++dydx) {
      int dy = dydx / 3, dx = dydx % 3;
      int y = h + dy - 1, xx = w + dx - 1;
      bool ok = (y >= 0) & (y < Hh) & (xx >= 0) & (xx < Ww);
      const float* xp = xb + y * Ww + xx;
      int c0 = s * 16;
#pragma unroll
      for (int i = 0; i < 16; ++i) {
        int c = c0 + i;
        float v = ok ? xp[(size_t)c * (Hh * Ww)] : 0.f;
        vrow[c * 9 + dydx] = f2bf(v);
      }
    }
  }
  __syncthreads();

  // ---- phase 2: WMMA GEMM  D[32 x 32] = W2[32 x 576] * V^T[576 x 32] ----
  int lane  = tid & 31;
  int wv    = tid >> 5;              // 0..3
  int mtile = wv >> 1;               // 0..1
  int ntile = wv & 1;                // 0..1
  int nloc  = lane & 15;
  int hi16  = lane >> 4;
  const unsigned short* arow = wb2 + (size_t)(16 * mtile + nloc) * KDIM + (hi16 << 3);
  const unsigned short* bp   = &Vlds[(16 * ntile + nloc) * LDSTR + (hi16 << 3)];

  v8f acc = {};
  union ABu { v16bf v; uint4 q[2]; };
  for (int kb = 0; kb < KDIM; kb += 32) {
    ABu a, bb;
    a.q[0]  = *(const uint4*)(arow + kb);
    a.q[1]  = *(const uint4*)(arow + kb + 16);
    bb.q[0] = *(const uint4*)(bp + kb);
    bb.q[1] = *(const uint4*)(bp + kb + 16);
    acc = __builtin_amdgcn_wmma_f32_16x16x32_bf16(false, a.v, false, bb.v,
                                                  (short)0, acc, false, false);
  }

  // ---- phase 3: bias + (sigmoid for mask rows), scatter to offs / maskb ----
  int p_local = 16 * ntile + nloc;
  int g = pix0 + p_local;
  int b = g >> 14, hw = g & 16383;
#pragma unroll
  for (int r = 0; r < 8; ++r) {
    int m = 16 * mtile + 8 * hi16 + r;
    float v = acc[r];
    if (m < 18) {
      offs[((size_t)(b * 18 + m) << 14) + hw] = v + ob[m];
    } else if (m < 27) {
      float z = v + mb[m - 18];
      maskb[((size_t)(b * 9 + (m - 18)) << 14) + hw] = 1.f / (1.f + expf(-z));
    }
  }
}

// ---------------- k2: fused bilinear sampling + bf16 WMMA GEMM ----------------
// Tile: 128 Cout x 32 pixels per workgroup; 8 waves; wave w -> rows [16w,16w+16).
__global__ void __launch_bounds__(256) k_deform_gemm(
    const float* __restrict__ x, const float* __restrict__ offs,
    const float* __restrict__ maskb, const unsigned short* __restrict__ wb,
    float* __restrict__ out) {
  __shared__ unsigned short Vlds[TN * LDSTR];

  int tid  = threadIdx.x;
  int pix0 = blockIdx.x * TN;

  // ---- phase 1: sample 32 pixels x 576 K-values into LDS (bf16) ----
  {
    int p = tid >> 3;            // 0..31 pixel in tile
    int s = tid & 7;             // 8 threads/pixel -> 8 channels each
    int g = pix0 + p;
    int b = g >> 14, hw = g & 16383;
    int h = hw >> 7, w = hw & 127;
    const float* xb = x + (size_t)b * CIN * (Hh * Ww);
    for (int kk = 0; kk < 9; ++kk) {
      float oy = offs[((size_t)(b * 18 + 2 * kk) << 14) + hw];
      float ox = offs[((size_t)(b * 18 + 2 * kk + 1) << 14) + hw];
      float mk = maskb[((size_t)(b * 9 + kk) << 14) + hw];
      float py = oy + (float)(kk / 3) + (float)(h - 1);
      float px = ox + (float)(kk % 3) + (float)(w - 1);
      float y0f = floorf(py), x0f = floorf(px);
      float wy1 = py - y0f, wx1 = px - x0f;
      float wy0 = 1.f - wy1, wx0 = 1.f - wx1;
      int y0 = (int)y0f, x0 = (int)x0f;
      int y1 = y0 + 1, x1 = x0 + 1;
      bool vy0 = (y0 >= 0) & (y0 < Hh), vy1 = (y1 >= 0) & (y1 < Hh);
      bool vx0 = (x0 >= 0) & (x0 < Ww), vx1 = (x1 >= 0) & (x1 < Ww);
      float w00 = (vy0 && vx0) ? wy0 * wx0 : 0.f;
      float w01 = (vy0 && vx1) ? wy0 * wx1 : 0.f;
      float w10 = (vy1 && vx0) ? wy1 * wx0 : 0.f;
      float w11 = (vy1 && vx1) ? wy1 * wx1 : 0.f;
      int yc0 = min(max(y0, 0), Hh - 1), yc1 = min(max(y1, 0), Hh - 1);
      int xc0 = min(max(x0, 0), Ww - 1), xc1 = min(max(x1, 0), Ww - 1);
      int i00 = yc0 * Ww + xc0, i01 = yc0 * Ww + xc1;
      int i10 = yc1 * Ww + xc0, i11 = yc1 * Ww + xc1;
      int c0 = s * 8;
      const float* xc = xb + (size_t)c0 * (Hh * Ww);
      unsigned short* vrow = &Vlds[p * LDSTR + c0 * 9 + kk];
#pragma unroll
      for (int i = 0; i < 8; ++i) {
        const float* xp = xc + (size_t)i * (Hh * Ww);
        float v = w00 * xp[i00] + w01 * xp[i01] + w10 * xp[i10] + w11 * xp[i11];
        vrow[i * 9] = f2bf(v * mk);
      }
    }
  }
  __syncthreads();

  // ---- phase 2: WMMA GEMM  D[128 x 32] = W[128 x 576] * V^T[576 x 32] ----
  int lane = tid & 31;
  int wv   = tid >> 5;                 // wave id 0..7
  int nloc = lane & 15;
  int hi16 = lane >> 4;                // 0 or 1
  const unsigned short* arow = wb + (size_t)(16 * wv + nloc) * KDIM + (hi16 << 3);
  const unsigned short* b0p = &Vlds[nloc * LDSTR + (hi16 << 3)];
  const unsigned short* b1p = &Vlds[(nloc + 16) * LDSTR + (hi16 << 3)];

  v8f acc0 = {}; v8f acc1 = {};
  union ABu { v16bf v; uint4 q[2]; };

  for (int kb = 0; kb < KDIM; kb += 32) {
    ABu a, b0, b1;
    a.q[0]  = *(const uint4*)(arow + kb);
    a.q[1]  = *(const uint4*)(arow + kb + 16);
    b0.q[0] = *(const uint4*)(b0p + kb);
    b0.q[1] = *(const uint4*)(b0p + kb + 16);
    b1.q[0] = *(const uint4*)(b1p + kb);
    b1.q[1] = *(const uint4*)(b1p + kb + 16);
    acc0 = __builtin_amdgcn_wmma_f32_16x16x32_bf16(false, a.v, false, b0.v,
                                                   (short)0, acc0, false, false);
    acc1 = __builtin_amdgcn_wmma_f32_16x16x32_bf16(false, a.v, false, b1.v,
                                                   (short)0, acc1, false, false);
  }

  // ---- write raw output: out[b][o][hw], o = 16*wv + 8*hi16 + r ----
#pragma unroll
  for (int t = 0; t < 2; ++t) {
    v8f a = t ? acc1 : acc0;
    int p_local = 16 * t + nloc;
    int g = pix0 + p_local;
    int b = g >> 14, hw = g & 16383;
    float* op = out + ((size_t)(b * COUT + 16 * wv + 8 * hi16) << 14) + hw;
#pragma unroll
    for (int r = 0; r < 8; ++r) op[(size_t)r << 14] = a[r];
  }
}

// ---------------- k3: per-channel sum / sumsq (one block per channel) ----------------
__global__ void __launch_bounds__(256) k_bnstats(
    const float* __restrict__ out, float* __restrict__ sums, float* __restrict__ sumsq) {
  __shared__ float r1[256], r2[256];
  int o = blockIdx.x, tid = threadIdx.x;
  float s = 0.f, s2 = 0.f;
  for (int n = tid; n < NPIX; n += 256) {
    int b = n >> 14, hw = n & 16383;
    float v = out[((size_t)(b * COUT + o) << 14) + hw];
    s += v; s2 += v * v;
  }
  r1[tid] = s; r2[tid] = s2;
  __syncthreads();
  for (int st = 128; st > 0; st >>= 1) {
    if (tid < st) { r1[tid] += r1[tid + st]; r2[tid] += r2[tid + st]; }
    __syncthreads();
  }
  if (tid == 0) { sums[o] = r1[0]; sumsq[o] = r2[0]; }
}

// ---------------- k4: BN normalize + ReLU in place ----------------
__global__ void __launch_bounds__(256) k_bnrelu(
    float* __restrict__ out, const float* __restrict__ sums,
    const float* __restrict__ sumsq) {
  int bo = blockIdx.x;                 // b*128 + o
  int o = bo & 127;
  const float invN = 1.f / (float)NPIX;
  float mean = sums[o] * invN;
  float var  = sumsq[o] * invN - mean * mean;
  float inv  = rsqrtf(var + 1e-5f);
  size_t base = (size_t)bo << 14;
  for (int i = threadIdx.x; i < 16384; i += 256) {
    float v = (out[base + i] - mean) * inv;
    out[base + i] = fmaxf(v, 0.f);
  }
}

// ---------------- launch ----------------
extern "C" void kernel_launch(void* const* d_in, const int* in_sizes, int n_in,
                              void* d_out, int out_size, void* d_ws, size_t ws_size,
                              hipStream_t stream) {
  const float* x  = (const float*)d_in[0];
  const float* ow = (const float*)d_in[1];
  const float* ob = (const float*)d_in[2];
  const float* mw = (const float*)d_in[3];
  const float* mb = (const float*)d_in[4];
  const float* cw = (const float*)d_in[5];
  float* out = (float*)d_out;
  char* ws = (char*)d_ws;

  // workspace layout
  unsigned short* wb  = (unsigned short*)(ws);               // 147456 B   main weights bf16
  float* offs  = (float*)(ws + 147456);                      // 9437184 B  offset NCHW fp32
  float* maskb = (float*)(ws + 9584640);                     // 4718592 B  mask NCHW fp32
  float* sums  = (float*)(ws + 14303232);                    // 512 B
  float* sumsq = (float*)(ws + 14303744);                    // 512 B
  unsigned short* wb2 = (unsigned short*)(ws + 14304256);    // 36864 B    offset+mask weights bf16 [32][576]

  k_cvt_w      <<<288, 256, 0, stream>>>(cw, wb);
  k_cvt_om     <<<72, 256, 0, stream>>>(ow, mw, wb2);
  k_conv_om    <<<NPIX / TN, 128, 0, stream>>>(x, wb2, ob, mb, offs, maskb);
  k_deform_gemm<<<NPIX / TN, 256, 0, stream>>>(x, offs, maskb, wb, out);
  k_bnstats    <<<COUT, 256, 0, stream>>>(out, sums, sumsq);
  k_bnrelu     <<<(NPIX / 16384) * COUT, 256, 0, stream>>>(out, sums, sumsq);
}
